// PCWrapper_25752623907132
// MI455X (gfx1250) — matrix operand order repaired and checked by reading, not compiled, not run on hardware
//
#include <hip/hip_runtime.h>
#include <hip/hip_bf16.h>

// ---------------------------------------------------------------------------
// Problem constants (from the reference)
// ---------------------------------------------------------------------------
#define B_SZ    2048
#define IMG     784          // 28*28
#define IN_LP   832          // 32*26
#define IN_NET  1568         // 2*784
#define NDIR    32
#define OUT_NET 25088        // NDIR*IMG

// padded sizes for the async (preconv) path: no in-kernel tails needed
#define KP_NET  1600         // 1568 -> 25 * 64
#define NP_LP   896          // 784  -> 7  * 128

typedef __attribute__((ext_vector_type(16))) __bf16 v16bf;
typedef __attribute__((ext_vector_type(8)))  float  v8f;

// fp32 -> bf16, round-to-nearest-even
static __device__ __forceinline__ __bf16 f2bf(float f) {
  unsigned u = __builtin_bit_cast(unsigned, f);
  unsigned r = u + 0x7FFFu + ((u >> 16) & 1u);
  unsigned short h = (unsigned short)(r >> 16);
  return __builtin_bit_cast(__bf16, h);
}

// low 32 bits of a flat pointer to LDS == LDS byte offset (aperture in high bits)
static __device__ __forceinline__ unsigned lds32(const void* p) {
  return (unsigned)(unsigned long long)(uintptr_t)p;
}
// CDNA5 async DMA: LDS[vdst] = MEM[vaddr], 16 B, tracked by ASYNCcnt
static __device__ __forceinline__ void async_b128(unsigned lds, unsigned long long ga) {
  asm volatile("global_load_async_to_lds_b128 %0, %1, off"
               :: "v"(lds), "v"(ga) : "memory");
}

// ---------------------------------------------------------------------------
// Strided fp32 -> bf16 convert with zero padding (rows beyond src_rows and
// cols beyond src_cols become 0).  dst is row-major [*, dst_cols], written in
// 8-element chunks (dst_cols % 8 == 0, src_cols % 8 == 0 so chunks never
// straddle the valid/pad boundary).
// ---------------------------------------------------------------------------
__global__ __launch_bounds__(256) void cvt_pad_kernel(
    const float* __restrict__ src, __bf16* __restrict__ dst,
    int src_rows, int src_cols, int dst_cols, long n) {
  long i = ((long)blockIdx.x * 256 + threadIdx.x) * 8;
  if (i >= n) return;
  int row = (int)(i / dst_cols);
  int col = (int)(i - (long)row * dst_cols);
  union { __bf16 h[8]; uint4 q; } o;
  if (row < src_rows && col < src_cols) {
    const float* s = src + (size_t)row * src_cols + col;
    float4 a = ((const float4*)s)[0];
    float4 b = ((const float4*)s)[1];
    o.h[0]=f2bf(a.x); o.h[1]=f2bf(a.y); o.h[2]=f2bf(a.z); o.h[3]=f2bf(a.w);
    o.h[4]=f2bf(b.x); o.h[5]=f2bf(b.y); o.h[6]=f2bf(b.z); o.h[7]=f2bf(b.w);
  } else {
    o.q = uint4{0u, 0u, 0u, 0u};
  }
  *(uint4*)(dst + i) = o.q;
}

// ---------------------------------------------------------------------------
// Fill columns [784, ld) of the bf16 activation matrix Xn[B][ld]:
//   col in [784,1568): (x_restored - 0.5)/0.2 ; col >= 1568: 0 (K padding)
// ---------------------------------------------------------------------------
__global__ __launch_bounds__(256) void norm_restored_kernel(
    const float* __restrict__ xr, __bf16* __restrict__ Xn, int ld) {
  int cols = ld - IMG;
  int i = blockIdx.x * 256 + threadIdx.x;
  if (i >= B_SZ * cols) return;
  int b = i / cols;
  int col = IMG + (i - b * cols);
  float v = (col < IN_NET) ? (xr[(size_t)b * IMG + (col - IMG)] - 0.5f) * 5.0f : 0.0f;
  Xn[(size_t)b * ld + col] = f2bf(v);
}

// ---------------------------------------------------------------------------
// ASYNC double-buffered WMMA GEMM (preconv path; operands bf16, N % 128 == 0,
// K % 64 == 0 — guaranteed by workspace padding).
//   D = epilogue(A[M,K] @ B[N,K]^T + bias)
//   EPI==0: D bf16, val = (acc + bias - 0.5) * 5      (GEMM1 -> activations)
//   EPI==1: D fp32, val = (acc + bias) * 0.2          (GEMM2 -> w_mat)
//
// Block tile 64(M) x 128(N) x 64(K), 8 wave32s.  Each thread issues exactly
// 6 global_load_async_to_lds_b128 per slab (uniform => compile-time
// s_wait_asynccnt immediates are safe).  Slab s+1 streams into buffer s^1
// while buffer s feeds 8 back-to-back v_wmma_f32_16x16x32_bf16; async loads
// complete in order, so s_wait_asynccnt 6 <=> current slab landed.
// LDS rows padded to 72 halves so 16-lane b128 fragment reads are
// bank-conflict free.
// ---------------------------------------------------------------------------
template<int EPI>
__global__ __launch_bounds__(256) void wmma_gemm_async_kernel(
    const __bf16* __restrict__ Aptr, int lda,
    const __bf16* __restrict__ Bptr, int ldb,
    const float* __restrict__ bias,
    void* __restrict__ Dptr, int ldd, int Nstore, int K)
{
  constexpr int KT = 64;
  constexpr int LK = KT + 8;        // 72 halves (144 B) row stride
  __shared__ __align__(16) __bf16 As[2][64][LK];
  __shared__ __align__(16) __bf16 Bs[2][128][LK];

  const int m0   = blockIdx.y * 64;
  const int n0   = blockIdx.x * 128;
  const int t    = threadIdx.x;
  const int lane = t & 31;
  const int wave = t >> 5;
  const int mi   = wave >> 1;       // 0..3  (16-row m-tile)
  const int nq   = wave & 1;        // 0..1  (64-col n-slab)

  v8f acc[4];
  #pragma unroll
  for (int i = 0; i < 4; ++i)
    #pragma unroll
    for (int e = 0; e < 8; ++e) acc[i][e] = 0.0f;

  // staging coordinates: A 64x64 (2 x 16B per thread), B 128x64 (4 x 16B)
  const int ar  = t >> 2;           // 0..63
  const int ac  = (t & 3) * 16;     // 0,16,32,48
  const int br  = t >> 1;           // 0..127
  const int bcc = (t & 1) * 32;     // 0,32

  const unsigned long long gA =
      (unsigned long long)(uintptr_t)(Aptr + (size_t)(m0 + ar) * lda + ac);
  const unsigned long long gB =
      (unsigned long long)(uintptr_t)(Bptr + (size_t)(n0 + br) * ldb + bcc);
  const unsigned lA[2] = { lds32(&As[0][ar][ac]), lds32(&As[1][ar][ac]) };
  const unsigned lB[2] = { lds32(&Bs[0][br][bcc]), lds32(&Bs[1][br][bcc]) };

  auto stage = [&](int buf, int k0) {
    unsigned long long ga = gA + (unsigned long long)k0 * 2;
    unsigned long long gb = gB + (unsigned long long)k0 * 2;
    async_b128(lA[buf],      ga);
    async_b128(lA[buf] + 16, ga + 16);
    async_b128(lB[buf],      gb);
    async_b128(lB[buf] + 16, gb + 16);
    async_b128(lB[buf] + 32, gb + 32);
    async_b128(lB[buf] + 48, gb + 48);
  };

  const int nslabs = K / KT;
  stage(0, 0);

  const int khalf = lane >> 4;
  const int lr    = lane & 15;
  const int am    = mi * 16 + lr;
  union Frag { uint4 q[2]; v16bf v; };

  for (int s = 0; s < nslabs; ++s) {
    const int cur = s & 1;
    if (s + 1 < nslabs) {
      stage(cur ^ 1, (s + 1) * KT);
      asm volatile("s_wait_asynccnt 6" ::: "memory");   // cur slab landed
    } else {
      asm volatile("s_wait_asynccnt 0" ::: "memory");
    }
    __syncthreads();   // all waves: cur slab visible

    // ---- fragments in contiguous k order, identical for A and B ----
    // A (16x32): lane L row = L&15; 16-half run at K = 16*(L>>4)
    // B (32x16): lane L col = L&15; 16-half run at K = 16*(L>>4)
    Frag fa[2];
    #pragma unroll
    for (int kk = 0; kk < 2; ++kk) {
      fa[kk].q[0] = *(const uint4*)&As[cur][am][kk * 32 + 16 * khalf];
      fa[kk].q[1] = *(const uint4*)&As[cur][am][kk * 32 + 16 * khalf + 8];
    }
    Frag fb[2][4];
    #pragma unroll
    for (int ti = 0; ti < 4; ++ti) {
      const int bn = (nq * 4 + ti) * 16 + lr;
      #pragma unroll
      for (int kk = 0; kk < 2; ++kk) {
        fb[kk][ti].q[0] = *(const uint4*)&Bs[cur][bn][kk * 32 + 16 * khalf];
        fb[kk][ti].q[1] = *(const uint4*)&Bs[cur][bn][kk * 32 + 16 * khalf + 8];
      }
    }
    #pragma unroll
    for (int kk = 0; kk < 2; ++kk)
      #pragma unroll
      for (int ti = 0; ti < 4; ++ti)
        acc[ti] = __builtin_amdgcn_wmma_f32_16x16x32_bf16(
            false, fa[kk].v, false, fb[kk][ti].v, (short)0, acc[ti], false, false);

    __syncthreads();   // all reads of cur done before slab s+2 overwrites it
  }

  // ---- epilogue: C/D layout — lane L: N = L&15, VGPR r: M = r + 8*(L>>4) ----
  const int mbase = m0 + mi * 16 + 8 * (lane >> 4);
  #pragma unroll
  for (int ti = 0; ti < 4; ++ti) {
    const int n = n0 + (nq * 4 + ti) * 16 + (lane & 15);
    if (n < Nstore) {
      const float bv = bias[n];
      #pragma unroll
      for (int r = 0; r < 8; ++r) {
        float v = acc[ti][r] + bv;
        if (EPI == 0) {
          ((__bf16*)Dptr)[(size_t)(mbase + r) * ldd + n] = f2bf((v - 0.5f) * 5.0f);
        } else {
          ((float*)Dptr)[(size_t)(mbase + r) * ldd + n] = v * 0.2f;
        }
      }
    }
  }
}

// ---------------------------------------------------------------------------
// Synchronous fallback WMMA GEMM (used when ws_size cannot hold the padded
// bf16 weight copies).  Converts fp32 operands while staging; guards N/K.
// ---------------------------------------------------------------------------
template<bool A_BF16, bool B_BF16, int EPI>
__global__ __launch_bounds__(256) void wmma_gemm_kernel(
    const void*  __restrict__ Aptr, int lda,
    const void*  __restrict__ Bptr, int ldb,
    const float* __restrict__ bias,
    void* __restrict__ Dptr, int ldd, int N, int K)
{
  constexpr int KT    = 64;
  constexpr int LDS_K = KT + 8;
  __shared__ __align__(16) __bf16 As[64][LDS_K];
  __shared__ __align__(16) __bf16 Bs[128][LDS_K];

  const int m0   = blockIdx.y * 64;
  const int n0   = blockIdx.x * 128;
  const int t    = threadIdx.x;
  const int lane = t & 31;
  const int wave = t >> 5;
  const int mi   = wave >> 1;
  const int nq   = wave & 1;

  v8f acc[4];
  #pragma unroll
  for (int i = 0; i < 4; ++i)
    #pragma unroll
    for (int e = 0; e < 8; ++e) acc[i][e] = 0.0f;

  const int ar  = t >> 2;
  const int ac  = (t & 3) * 16;
  const int br  = t >> 1;
  const int bcc = (t & 1) * 32;
  const int gn  = n0 + br;

  const uint4 z4 = {0u, 0u, 0u, 0u};

  for (int k0 = 0; k0 < K; k0 += KT) {
    __syncthreads();
    {
      __bf16* dst = &As[ar][ac];
      if (k0 + ac < K) {
        if (A_BF16) {
          const __bf16* src = (const __bf16*)Aptr + (size_t)(m0 + ar) * lda + k0 + ac;
          ((uint4*)dst)[0] = ((const uint4*)src)[0];
          ((uint4*)dst)[1] = ((const uint4*)src)[1];
        } else {
          const float* src = (const float*)Aptr + (size_t)(m0 + ar) * lda + k0 + ac;
          #pragma unroll
          for (int c = 0; c < 4; ++c) {
            float4 f = ((const float4*)src)[c];
            dst[4*c+0]=f2bf(f.x); dst[4*c+1]=f2bf(f.y);
            dst[4*c+2]=f2bf(f.z); dst[4*c+3]=f2bf(f.w);
          }
        }
      } else {
        ((uint4*)dst)[0] = z4;
        ((uint4*)dst)[1] = z4;
      }
    }
    {
      __bf16* dst = &Bs[br][bcc];
      if (gn < N && (k0 + bcc) < K) {
        if (B_BF16) {
          const __bf16* src = (const __bf16*)Bptr + (size_t)gn * ldb + k0 + bcc;
          #pragma unroll
          for (int c = 0; c < 4; ++c) ((uint4*)dst)[c] = ((const uint4*)src)[c];
        } else {
          const float* src = (const float*)Bptr + (size_t)gn * ldb + k0 + bcc;
          if (k0 + KT < K) __builtin_prefetch(src + KT, 0, 1);
          #pragma unroll
          for (int c = 0; c < 8; ++c) {
            float4 f = ((const float4*)src)[c];
            dst[4*c+0]=f2bf(f.x); dst[4*c+1]=f2bf(f.y);
            dst[4*c+2]=f2bf(f.z); dst[4*c+3]=f2bf(f.w);
          }
        }
      } else {
        #pragma unroll
        for (int c = 0; c < 4; ++c) ((uint4*)dst)[c] = z4;
      }
    }
    __syncthreads();

    const int khalf = lane >> 4;
    const int lr    = lane & 15;
    const int am    = mi * 16 + lr;
    union Frag { uint4 q[2]; v16bf v; };

    Frag fa[2];
    #pragma unroll
    for (int kk = 0; kk < 2; ++kk) {
      fa[kk].q[0] = *(const uint4*)&As[am][kk * 32 + 16 * khalf];
      fa[kk].q[1] = *(const uint4*)&As[am][kk * 32 + 16 * khalf + 8];
    }
    Frag fb[2][4];
    #pragma unroll
    for (int ti = 0; ti < 4; ++ti) {
      const int bn = (nq * 4 + ti) * 16 + lr;
      #pragma unroll
      for (int kk = 0; kk < 2; ++kk) {
        fb[kk][ti].q[0] = *(const uint4*)&Bs[bn][kk * 32 + 16 * khalf];
        fb[kk][ti].q[1] = *(const uint4*)&Bs[bn][kk * 32 + 16 * khalf + 8];
      }
    }
    #pragma unroll
    for (int kk = 0; kk < 2; ++kk)
      #pragma unroll
      for (int ti = 0; ti < 4; ++ti)
        acc[ti] = __builtin_amdgcn_wmma_f32_16x16x32_bf16(
            false, fa[kk].v, false, fb[kk][ti].v, (short)0, acc[ti], false, false);
  }

  const int mbase = m0 + mi * 16 + 8 * (lane >> 4);
  #pragma unroll
  for (int ti = 0; ti < 4; ++ti) {
    const int n = n0 + (nq * 4 + ti) * 16 + (lane & 15);
    if (n < N) {
      const float bv = bias[n];
      #pragma unroll
      for (int r = 0; r < 8; ++r) {
        float v = acc[ti][r] + bv;
        if (EPI == 0) {
          ((__bf16*)Dptr)[(size_t)(mbase + r) * ldd + n] = f2bf((v - 0.5f) * 5.0f);
        } else {
          ((float*)Dptr)[(size_t)(mbase + r) * ldd + n] = v * 0.2f;
        }
      }
    }
  }
}

// ---------------------------------------------------------------------------
// Modified Gram-Schmidt, one workgroup (256 thr = 8 wave32) per batch row,
// in place on d_out.  Basis lives in dynamic LDS (32*784 floats = 98 KB,
// fits the 320 KB WGP LDS).  Sequential projections with normalized basis;
// output rows are the projected-but-unnormalized vectors (reference semantics).
// ---------------------------------------------------------------------------
static __device__ __forceinline__ float block_reduce_256(float v, float* red) {
  #pragma unroll
  for (int off = 16; off > 0; off >>= 1) v += __shfl_xor(v, off, 32);
  const int wid  = threadIdx.x >> 5;
  const int lane = threadIdx.x & 31;
  if (lane == 0) red[wid] = v;
  __syncthreads();
  if (wid == 0) {
    float s = (lane < 8) ? red[lane] : 0.0f;
    #pragma unroll
    for (int off = 4; off > 0; off >>= 1) s += __shfl_xor(s, off, 32);
    if (lane == 0) red[8] = s;
  }
  __syncthreads();
  float r = red[8];
  __syncthreads();
  return r;
}

__global__ __launch_bounds__(256) void gram_schmidt_kernel(float* __restrict__ out) {
  extern __shared__ float basis[];          // [NDIR][IMG]
  __shared__ float red[9];
  const int t = threadIdx.x;
  float* w_mat = out + (size_t)blockIdx.x * (NDIR * IMG);

  for (int i = 0; i < NDIR; ++i) {
    float w[4];
    #pragma unroll
    for (int k = 0; k < 4; ++k) {
      int idx = t + k * 256;
      w[k] = (idx < IMG) ? w_mat[i * IMG + idx] : 0.0f;
    }
    for (int j = 0; j < i; ++j) {
      float p = 0.0f;
      #pragma unroll
      for (int k = 0; k < 4; ++k) {
        int idx = t + k * 256;
        if (idx < IMG) p += w[k] * basis[j * IMG + idx];
      }
      float dot = block_reduce_256(p, red);
      #pragma unroll
      for (int k = 0; k < 4; ++k) {
        int idx = t + k * 256;
        if (idx < IMG) w[k] -= dot * basis[j * IMG + idx];
      }
    }
    float p = 0.0f;
    #pragma unroll
    for (int k = 0; k < 4; ++k) p += w[k] * w[k];
    float nrm2 = block_reduce_256(p, red);
    float inv  = rsqrtf(nrm2);
    #pragma unroll
    for (int k = 0; k < 4; ++k) {
      int idx = t + k * 256;
      if (idx < IMG) {
        w_mat[i * IMG + idx] = w[k];            // x_orth row i (unnormalized)
        basis[i * IMG + idx] = w[k] * inv;      // normalized basis for later rows
      }
    }
    __syncthreads();
  }
}

// ---------------------------------------------------------------------------
// Launch
// ---------------------------------------------------------------------------
extern "C" void kernel_launch(void* const* d_in, const int* in_sizes, int n_in,
                              void* d_out, int out_size, void* d_ws, size_t ws_size,
                              hipStream_t stream) {
  const float* x_dist = (const float*)d_in[0];   // [2048, 832]
  const float* x_rest = (const float*)d_in[1];   // [2048, 1, 28, 28]
  const float* W_lp   = (const float*)d_in[2];   // [784, 832]
  const float* b_lp   = (const float*)d_in[3];   // [784]
  const float* W_net  = (const float*)d_in[4];   // [25088, 1568]
  const float* b_net  = (const float*)d_in[5];   // [25088]
  float* out = (float*)d_out;                    // [2048, 32, 784]

  // preconv workspace layout (bytes): all operands bf16, padded for the
  // async kernel (N mult of 128, K mult of 64)
  const size_t sz_xn   = (size_t)B_SZ  * KP_NET * 2;   // [2048][1600]
  const size_t sz_xd   = (size_t)B_SZ  * IN_LP  * 2;   // [2048][832]
  const size_t sz_wlp  = (size_t)NP_LP * IN_LP  * 2;   // [896][832]
  const size_t sz_wnet = (size_t)OUT_NET * KP_NET * 2; // [25088][1600]
  const size_t need_pre = sz_xn + sz_xd + sz_wlp + sz_wnet;   // ~92 MB

  const bool preconv = (ws_size >= need_pre);   // deterministic host choice

  if (preconv) {
    __bf16* Xn = (__bf16*)d_ws;
    __bf16* Xd = (__bf16*)((char*)d_ws + sz_xn);
    __bf16* Wl = (__bf16*)((char*)d_ws + sz_xn + sz_xd);
    __bf16* Wn = (__bf16*)((char*)d_ws + sz_xn + sz_xd + sz_wlp);

    // activations right half + K padding
    {
      int elems = B_SZ * (KP_NET - IMG);
      norm_restored_kernel<<<(elems + 255) / 256, 256, 0, stream>>>(x_rest, Xn, KP_NET);
    }
    // one-shot bf16 conversion + padding of GEMM operands
    {
      long n1 = (long)B_SZ * IN_LP;
      long n2 = (long)NP_LP * IN_LP;
      long n3 = (long)OUT_NET * KP_NET;
      cvt_pad_kernel<<<(int)((n1/8 + 255)/256), 256, 0, stream>>>(x_dist, Xd, B_SZ, IN_LP, IN_LP, n1);
      cvt_pad_kernel<<<(int)((n2/8 + 255)/256), 256, 0, stream>>>(W_lp,  Wl, IMG,  IN_LP, IN_LP, n2);
      cvt_pad_kernel<<<(int)((n3/8 + 255)/256), 256, 0, stream>>>(W_net, Wn, OUT_NET, IN_NET, KP_NET, n3);
    }
    // GEMM1: xd = x_distorted @ W_lp^T + b_lp, fused normalize -> bf16 left half
    {
      dim3 grid(NP_LP / 128, B_SZ / 64);           // (7, 32)
      wmma_gemm_async_kernel<0><<<grid, 256, 0, stream>>>(
          Xd, IN_LP, Wl, IN_LP, b_lp, (void*)Xn, KP_NET, IMG, IN_LP);
    }
    // GEMM2: w_mat = (Xn @ W_net^T + b_net) * 0.2 -> fp32 into d_out
    {
      dim3 grid(OUT_NET / 128, B_SZ / 64);         // (196, 32)
      wmma_gemm_async_kernel<1><<<grid, 256, 0, stream>>>(
          Xn, KP_NET, Wn, KP_NET, b_net, (void*)out, OUT_NET, OUT_NET, KP_NET);
    }
  } else {
    __bf16* Xn = (__bf16*)d_ws;                    // [2048][1568] only
    {
      int elems = B_SZ * (IN_NET - IMG);
      norm_restored_kernel<<<(elems + 255) / 256, 256, 0, stream>>>(x_rest, Xn, IN_NET);
    }
    {
      dim3 grid((IMG + 127) / 128, B_SZ / 64);
      wmma_gemm_kernel<false, false, 0><<<grid, 256, 0, stream>>>(
          (const void*)x_dist, IN_LP, (const void*)W_lp, IN_LP, b_lp,
          (void*)Xn, IN_NET, IMG, IN_LP);
    }
    {
      dim3 grid(OUT_NET / 128, B_SZ / 64);
      wmma_gemm_kernel<true, false, 1><<<grid, 256, 0, stream>>>(
          (const void*)Xn, IN_NET, (const void*)W_net, IN_NET, b_net,
          (void*)out, OUT_NET, OUT_NET, IN_NET);
    }
  }
  // Gram-Schmidt in place on d_out, one workgroup per batch row
  {
    size_t lds = (size_t)NDIR * IMG * sizeof(float);   // 100352 B
    gram_schmidt_kernel<<<B_SZ, 256, lds, stream>>>(out);
  }
}
